// SO3Conv_17678085390533
// MI455X (gfx1250) — matrix operand order repaired and checked by reading, not compiled, not run on hardware
//
#include <hip/hip_runtime.h>
#include <math.h>

typedef unsigned short u16;
typedef unsigned int   u32;

typedef __bf16 v16bf __attribute__((ext_vector_type(16)));
typedef float  v8f   __attribute__((ext_vector_type(8)));
typedef int    v4i   __attribute__((ext_vector_type(4)));

union FragU { v16bf v; uint4 q[2]; };

#if __has_builtin(__builtin_amdgcn_global_load_async_to_lds_b128) && \
    __has_builtin(__builtin_amdgcn_s_wait_asynccnt)
#define HAVE_ASYNC 1
#else
#define HAVE_ASYNC 0
#endif

// builtin signature (from hipcc diagnostic): (int4 as1* src, int4 as3* dst, Ii off, Ii cpol)
#define AS1P(p) ((__attribute__((address_space(1))) v4i*)(p))
#define AS3P(p) ((__attribute__((address_space(3))) v4i*)(p))

// round-to-nearest-even f32 -> bf16 bits
static __device__ __forceinline__ u16 f2bf(float f) {
    u32 x = __float_as_uint(f);
    u32 r = (x + 0x7FFFu + ((x >> 16) & 1u)) >> 16;
    return (u16)r;
}

#define B_IN        1024
#define F_IN        64
#define F_OUT       64
#define N_ROT       64
#define IRREP_FLAT  455

#define MT  256      // macro-tile rows  (M = 1024*d, exact multiple)
#define NT  64       // macro-tile cols  (N = 64*d, exact multiple)
#define KT  32       // k-step           (K = 64*d, exact multiple)
#define LDA 40       // LDS row stride in bf16 elems (80B: 16B-aligned, conflict-padded)

// ---------------------------------------------------------------------------
// Kernel 1: Bpack[n,k] = bf16( norm_l/8 * sum_n' D[n',i]*w[f,g,n'] ), pre-transposed
//           n = g*d + v, k = f*d + u, i = off + u*d + v.  Layout: row-major N x K.
// ---------------------------------------------------------------------------
__global__ __launch_bounds__(256) void so3_psi_pack(
    const float* __restrict__ D, const float* __restrict__ w,
    u16* __restrict__ Bpack, int d, int K, int off, float scale)
{
    int t = blockIdx.x * 256 + threadIdx.x;
    int total = K * K;
    if (t >= total) return;
    int nn = t / K, kk = t - nn * K;
    int f = kk / d, u = kk - f * d;
    int g = nn / d, v = nn - g * d;
    int i = off + u * d + v;
    const float* wrow = w + ((size_t)(f * F_OUT + g)) * N_ROT;
    float acc = 0.f;
    #pragma unroll 8
    for (int n = 0; n < N_ROT; ++n)
        acc += D[n * IRREP_FLAT + i] * wrow[n];
    Bpack[(size_t)nn * K + kk] = f2bf(acc * scale);
}

// ---------------------------------------------------------------------------
// Kernel 2: Apack[row=(b*d+m), k=(f*d+u)] = bf16( x[b, f, off + u*d + m] )
// ---------------------------------------------------------------------------
__global__ __launch_bounds__(256) void so3_a_pack(
    const float* __restrict__ x, u16* __restrict__ Apack,
    int d, int K, int off, int total)
{
    int t = blockIdx.x * 256 + threadIdx.x;
    if (t >= total) return;
    int row = t / K, kk = t - row * K;
    int b = row / d, m = row - b * d;
    int f = kk / d,  u = kk - f * d;
    float val = x[(size_t)(b * F_IN + f) * IRREP_FLAT + off + u * d + m];
    Apack[(size_t)row * K + kk] = f2bf(val);
}

// ---------------------------------------------------------------------------
// Kernel 3: per-l GEMM.  C[M x N] = A[M x K] * B[K x N], bf16 in / f32 out.
// Grid: (M/256, N/64).  8 waves, each owns a 32x64 C tile (2x4 v8f accums).
// Next-tile fill overlaps the 8 WMMAs: async global->LDS (ASYNCcnt) if the
// builtins exist, else a register-staged fetch/commit split.
// ---------------------------------------------------------------------------
__global__ __launch_bounds__(256) void so3_gemm(
    const u16* __restrict__ Ap, const u16* __restrict__ Bp,
    float* __restrict__ out, int d, int K, int off)
{
    __shared__ u16 sA[2][MT * LDA];   // 40 KB
    __shared__ u16 sB[2][NT * LDA];   // 10 KB

    const int t    = threadIdx.x;
    const int wave = t >> 5;
    const int lane = t & 31;
    const int rowBase = blockIdx.x * MT;
    const int nBase   = blockIdx.y * NT;

    // per-thread load coordinates (5 x 16B chunks per k-step)
    const int lr  = t >> 2;          // 0..63
    const int lck = (t & 3) * 8;     // 0/8/16/24

    const v8f zero = {0.f, 0.f, 0.f, 0.f, 0.f, 0.f, 0.f, 0.f};
    v8f acc[2][4];
    #pragma unroll
    for (int i = 0; i < 2; ++i)
        #pragma unroll
        for (int j = 0; j < 4; ++j) acc[i][j] = zero;

#if HAVE_ASYNC
    auto async_tile = [&](int buf, int kbase) {
        #pragma unroll
        for (int i = 0; i < 4; ++i) {
            int r = i * 64 + lr;
            __builtin_amdgcn_global_load_async_to_lds_b128(
                AS1P(Ap + (size_t)(rowBase + r) * K + kbase + lck),
                AS3P(&sA[buf][r * LDA + lck]), 0, 0);
        }
        __builtin_amdgcn_global_load_async_to_lds_b128(
            AS1P(Bp + (size_t)(nBase + lr) * K + kbase + lck),
            AS3P(&sB[buf][lr * LDA + lck]), 0, 0);
    };
#else
    uint4 ra[4], rb;
    auto fetch = [&](int kbase) {
        #pragma unroll
        for (int i = 0; i < 4; ++i) {
            int r = i * 64 + lr;
            ra[i] = *(const uint4*)(Ap + (size_t)(rowBase + r) * K + kbase + lck);
        }
        rb = *(const uint4*)(Bp + (size_t)(nBase + lr) * K + kbase + lck);
    };
    auto commit = [&](int buf) {
        #pragma unroll
        for (int i = 0; i < 4; ++i) {
            int r = i * 64 + lr;
            *(uint4*)(&sA[buf][r * LDA + lck]) = ra[i];
        }
        *(uint4*)(&sB[buf][lr * LDA + lck]) = rb;
    };
#endif

    const int rsel = lane & 15;
    const int hi   = (lane >> 4) & 1;
    const int hiA  = hi * 8;   // A: lanes 16-31 take K=8..15 / K=24..31 halves
    const int hiB  = hi * 16;  // unused: B uses hiA (same K order as A)

    const int nK = K / KT;

#if HAVE_ASYNC
    async_tile(0, 0);
    __builtin_amdgcn_s_wait_asynccnt(0);
    __syncthreads();
#else
    fetch(0);
    commit(0);
    __syncthreads();
#endif

    for (int kt = 0; kt < nK; ++kt) {
        const int cur = kt & 1;

#if HAVE_ASYNC
        if (kt + 1 < nK) async_tile(cur ^ 1, (kt + 1) * KT);
#else
        if (kt + 1 < nK) fetch((kt + 1) * KT);
#endif

        FragU a[2], b[4];
        #pragma unroll
        for (int i = 0; i < 2; ++i) {
            const u16* p = &sA[cur][(wave * 32 + i * 16 + rsel) * LDA];
            a[i].q[0] = *(const uint4*)(p + hiA);        // K 0..7  / 8..15
            a[i].q[1] = *(const uint4*)(p + hiA + 16);   // K 16..23 / 24..31
        }
        #pragma unroll
        for (int j = 0; j < 4; ++j) {
            const u16* p = &sB[cur][(j * 16 + rsel) * LDA];
            b[j].q[0] = *(const uint4*)(p + hiA);        // K 0..7  / 8..15
            b[j].q[1] = *(const uint4*)(p + hiA + 16);   // K 16..23 / 24..31
        }
        #pragma unroll
        for (int i = 0; i < 2; ++i)
            #pragma unroll
            for (int j = 0; j < 4; ++j)
                acc[i][j] = __builtin_amdgcn_wmma_f32_16x16x32_bf16(
                    false, a[i].v, false, b[j].v, (short)0, acc[i][j], false, false);

#if HAVE_ASYNC
        if (kt + 1 < nK) __builtin_amdgcn_s_wait_asynccnt(0);
#else
        if (kt + 1 < nK) commit(cur ^ 1);
#endif
        __syncthreads();
    }

    // Scatter store: row R=(b*d+m), col C=(g*d+v) -> y[b, g, off + v*d + m]
    #pragma unroll
    for (int j = 0; j < 4; ++j) {
        int cL = nBase + j * 16 + rsel;
        int g  = cL / d, v = cL - g * d;
        #pragma unroll
        for (int i = 0; i < 2; ++i) {
            #pragma unroll
            for (int p = 0; p < 8; ++p) {
                int rL = rowBase + wave * 32 + i * 16 + p + hi * 8;
                int bb = rL / d, m = rL - bb * d;
                out[(size_t)(bb * F_OUT + g) * IRREP_FLAT + off + v * d + m] = acc[i][j][p];
            }
        }
    }
}

// ---------------------------------------------------------------------------
extern "C" void kernel_launch(void* const* d_in, const int* in_sizes, int n_in,
                              void* d_out, int out_size, void* d_ws, size_t ws_size,
                              hipStream_t stream)
{
    const float* x = (const float*)d_in[0];   // (1024, 64, 455)
    const float* D = (const float*)d_in[1];   // (64, 455)
    const float* w = (const float*)d_in[2];   // (64, 64, 64)
    float* out = (float*)d_out;               // (1024, 64, 455)

    // workspace layout: Apack (bf16, 29,818,880 elems) then Bpack (bf16)
    static const size_t APACK_ELEMS = (size_t)65536 * 455;      // sum_l 65536*d^2
    u16* Apack = (u16*)d_ws;
    u16* Bpack = (u16*)((char*)d_ws + APACK_ELEMS * sizeof(u16));

    static const int offs[7] = {0, 1, 10, 35, 84, 165, 286};
    size_t aoff = 0, boff = 0;

    // pack passes
    for (int l = 0; l <= 6; ++l) {
        int d = 2 * l + 1;
        int K = 64 * d;
        float scale = 1.0f / (64.0f * sqrtf((float)d));  // 1/8 * 1/sqrt(64*d)
        int bTotal = K * K;
        so3_psi_pack<<<(bTotal + 255) / 256, 256, 0, stream>>>(
            D, w, Bpack + boff, d, K, offs[l], scale);
        int aTotal = B_IN * d * K;
        so3_a_pack<<<(aTotal + 255) / 256, 256, 0, stream>>>(
            x, Apack + aoff, d, K, offs[l], aTotal);
        aoff += (size_t)65536 * d * d;
        boff += (size_t)4096 * d * d;
    }

    // GEMM passes: M=1024d, N=64d, K=64d — all exact multiples of the tiles
    aoff = 0; boff = 0;
    for (int l = 0; l <= 6; ++l) {
        int d = 2 * l + 1;
        int K = 64 * d;
        dim3 grid(4 * d, d);
        so3_gemm<<<grid, 256, 0, stream>>>(
            Apack + aoff, Bpack + boff, out, d, K, offs[l]);
        aoff += (size_t)65536 * d * d;
        boff += (size_t)4096 * d * d;
    }
}
